// HeteroTransportCell_38697655337479
// MI455X (gfx1250) — matrix operand order repaired and checked by reading, not compiled, not run on hardware
//
#include <hip/hip_runtime.h>

// ---------------------------------------------------------------------------
// HeteroTransportCell for MI455X (gfx1250, wave32, WMMA).
// - se.fc2 (128x128 per edge, naive FLOP hotspot) folded into bw (128->1):
//   bw·(W2·r + b2) = (bw·W2)·r + bw·b2  -> precomputed 128-vector.
// - All dense GEMMs via v_wmma_f32_16x16x32_f16.
// - Tile staging fully vectorized: float4 global loads -> v8h ds_store_b128
//   (previous round compiled staging to per-element predicated loads with a
//   s_wait_loadcnt 0x0 stall per element).
// - Gate activations stored f16 to halve per-edge gather bytes (23.3 TB/s HBM
//   bound after the algebraic fusion).
// ---------------------------------------------------------------------------

typedef _Float16 half_t;
typedef __attribute__((ext_vector_type(16))) _Float16 v16h;
typedef __attribute__((ext_vector_type(8)))  _Float16 v8h;
typedef __attribute__((ext_vector_type(8)))  float    v8f;

#define N1_C 20000
#define N2_C 40000
#define E_OO_C 320000
#define E_TT_C 640000
#define E_OT_C 320000
#define E_TO_C 320000

__device__ inline v8f wmma32(v16h a, v16h b, v8f c) {
  return __builtin_amdgcn_wmma_f32_16x16x32_f16(false, a, false, b, (short)0, c, false, false);
}

// A-fragment: 16x32 f16 tile, row-major LDS [rows][ldk]. Lane l: row=l&15,
// K = kh..kh+7 and 16+kh..16+kh+7 where kh = (l>>4)*8  (CDNA5 ISA 7.12.2).
__device__ inline v16h load_a_frag(const half_t* base, int row0, int ldk, int lane) {
  int r  = row0 + (lane & 15);
  int kh = (lane >> 4) * 8;
  const half_t* p = base + r * ldk + kh;
  v8h lo = *(const v8h*)(p);
  v8h hi = *(const v8h*)(p + 16);
  v16h out;
#pragma unroll
  for (int i = 0; i < 8; ++i) { out[i] = lo[i]; out[i + 8] = hi[i]; }
  return out;
}

// B-fragment: B[k][n] = W[n][k]; lane l holds column n=l&15 with
// K = kb..kb+7 and 16+kb..16+kb+7 where kb = (l>>4)*8 (matches A). W tile row-major [n][k].
__device__ inline v16h load_b_frag(const half_t* base, int row0, int ldk, int lane) {
  int r  = row0 + (lane & 15);
  int kb = (lane >> 4) * 8;
  const half_t* p = base + r * ldk + kb;
  v8h lo = *(const v8h*)(p);
  v8h hi = *(const v8h*)(p + 16);
  v16h out;
#pragma unroll
  for (int i = 0; i < 8; ++i) { out[i] = lo[i]; out[i + 8] = hi[i]; }
  return out;
}

__device__ inline v8h cvt8(float4 a, float4 b) {
  v8h h;
  h[0] = (half_t)a.x; h[1] = (half_t)a.y; h[2] = (half_t)a.z; h[3] = (half_t)a.w;
  h[4] = (half_t)b.x; h[5] = (half_t)b.y; h[6] = (half_t)b.z; h[7] = (half_t)b.w;
  return h;
}

// ---------------------------------------------------------------------------
// Generic WMMA GEMM:  out[M,Nout] = act(A[M,K] @ W[:, wcol0:wcol0+K]^T + bias)
// K and lda/ldw are multiples of 8 for all call sites, so each 8-wide K
// segment is either fully valid or fully zero -> branch-free vector staging.
// ---------------------------------------------------------------------------
#define GT_LDK 40

__global__ __launch_bounds__(256)
void gemm_kernel(const float* __restrict__ A, int M, int K, int lda,
                 const float* __restrict__ W, int ldw, int wcol0,
                 const float* __restrict__ bias,
                 float* __restrict__ C, half_t* __restrict__ Ch,
                 int ldc, int ccol0, int Nout, int act, int accum, int wal)
{
  __shared__ __align__(16) half_t As[64 * GT_LDK];
  __shared__ __align__(16) half_t Ws[64 * GT_LDK];
  const int m0 = blockIdx.x * 64;
  const int n0 = blockIdx.y * 64;
  const int tid = threadIdx.x;
  const int lane = tid & 31, wave = tid >> 5;
  const int wm = (wave >> 1) * 16;      // row sub-tile
  const int wn = (wave & 1) * 32;       // col sub-tile (2 x 16)
  const int sr  = tid >> 2;             // staging row 0..63
  const int sk  = (tid & 3) * 8;        // staging K segment
  v8f acc0 = {}; v8f acc1 = {};

  for (int kb = 0; kb < K; kb += 32) {
    // A tile: one thread = one (row, 8-seg): 2x float4 load + 1x ds_store_b128
    {
      int gm = m0 + sr, gk = kb + sk;
      v8h h;
      if (gm < M && gk + 8 <= K) {
        const float* ap = A + (size_t)gm * lda + gk;
        h = cvt8(*(const float4*)ap, *(const float4*)(ap + 4));
      } else {
#pragma unroll
        for (int i = 0; i < 8; ++i) h[i] = (half_t)0.f;
      }
      *(v8h*)(&As[sr * GT_LDK + sk]) = h;
    }
    // W tile
    {
      int gn = n0 + sr, gk = kb + sk;
      v8h h;
      if (gn < Nout && gk + 8 <= K) {
        const float* wp = W + (size_t)gn * ldw + wcol0 + gk;
        if (wal) {
          h = cvt8(*(const float4*)wp, *(const float4*)(wp + 4));
        } else {
#pragma unroll
          for (int i = 0; i < 8; ++i) h[i] = (half_t)wp[i];
        }
      } else {
#pragma unroll
        for (int i = 0; i < 8; ++i) h[i] = (half_t)0.f;
      }
      *(v8h*)(&Ws[sr * GT_LDK + sk]) = h;
    }
    __syncthreads();
    v16h a  = load_a_frag(As, wm, GT_LDK, lane);
    v16h b0 = load_b_frag(Ws, wn, GT_LDK, lane);
    v16h b1 = load_b_frag(Ws, wn + 16, GT_LDK, lane);
    acc0 = wmma32(a, b0, acc0);
    acc1 = wmma32(a, b1, acc1);
    __syncthreads();
  }

  int col0 = n0 + wn + (lane & 15);
  int col1 = col0 + 16;
  int mb   = m0 + wm + (lane >> 4) * 8;
  float bias0 = 0.f, bias1 = 0.f;
  if (bias) {
    if (col0 < Nout) bias0 = bias[col0];
    if (col1 < Nout) bias1 = bias[col1];
  }
#pragma unroll
  for (int j = 0; j < 8; ++j) {
    int gm = mb + j;
    if (gm >= M) continue;
    if (col0 < Nout) {
      float x = acc0[j] + bias0;
      if (act == 1) x = fmaxf(x, 0.f);
      size_t o = (size_t)gm * ldc + ccol0 + col0;
      if (accum && C) x += C[o];
      if (C)  C[o]  = x;
      if (Ch) Ch[o] = (half_t)x;
    }
    if (col1 < Nout) {
      float x = acc1[j] + bias1;
      if (act == 1) x = fmaxf(x, 0.f);
      size_t o = (size_t)gm * ldc + ccol0 + col1;
      if (accum && C) x += C[o];
      if (C)  C[o]  = x;
      if (Ch) Ch[o] = (half_t)x;
    }
  }
}

// ---------------------------------------------------------------------------
// sd prep: cw[k] = sum_n bw[n]*W2[n][k];  cw[128] = bw·b2 + bw_b  (fc2 folding)
// ---------------------------------------------------------------------------
__global__ void sd_prep_kernel(const float* __restrict__ bw_w, const float* __restrict__ bw_b,
                               const float* __restrict__ W2, const float* __restrict__ b2,
                               float* __restrict__ cw)
{
  int k = threadIdx.x;
  float acc = 0.f;
  for (int n = 0; n < 128; ++n) acc = fmaf(bw_w[n], W2[n * 128 + k], acc);
  cw[k] = acc;
  if (k == 0) {
    float cb = bw_b[0];
    for (int n = 0; n < 128; ++n) cb = fmaf(bw_w[n], b2[n], cb);
    cw[128] = cb;
  }
}

// ---------------------------------------------------------------------------
// sd edge kernel: 128 edges/block. WMMA fc1 (K=16 padded to 32) with the
// collapsed bw·W2 dot folded into the epilogue; then dynamic gate (f16 v8h
// gathers) and 64-wide atomic scatter  msg[dst] += b_e*g_e*v[src].
// ---------------------------------------------------------------------------
__global__ __launch_bounds__(256)
void sd_edge_kernel(const int* __restrict__ src, const int* __restrict__ dst, int E,
                    const float* __restrict__ ea,
                    const float* __restrict__ xs_s, const float* __restrict__ xs_d,
                    const float* __restrict__ W1, const float* __restrict__ b1,
                    const float* __restrict__ cw,
                    const half_t* __restrict__ gsrc, const half_t* __restrict__ gdst,
                    const float* __restrict__ dgw, const float* __restrict__ dgb,
                    const float* __restrict__ v, float* __restrict__ msg, int wal)
{
  __shared__ __align__(16) half_t SC[128 * GT_LDK];
  __shared__ __align__(16) half_t WS[128 * GT_LDK];
  __shared__ float B1s[128], CWs[128], DGWs[128];
  __shared__ float BE[128], COEF[128];
  const int tid = threadIdx.x;
  const int lane = tid & 31, wave = tid >> 5;
  const int e0 = blockIdx.x * 128;

  // W1 tile: 128 rows x 32 (K padded 16->32); 512 segments / 256 threads
#pragma unroll
  for (int t2 = 0; t2 < 2; ++t2) {
    int idx = tid + t2 * 256;
    int r = idx >> 2, k = (idx & 3) * 8;
    v8h h;
    if (k < 16) {
      const float* wp = W1 + r * 16 + k;
      if (wal) {
        h = cvt8(*(const float4*)wp, *(const float4*)(wp + 4));
      } else {
#pragma unroll
        for (int i = 0; i < 8; ++i) h[i] = (half_t)wp[i];
      }
    } else {
#pragma unroll
      for (int i = 0; i < 8; ++i) h[i] = (half_t)0.f;
    }
    *(v8h*)(&WS[r * GT_LDK + k]) = h;
  }
  if (tid < 128) { B1s[tid] = b1[tid]; CWs[tid] = cw[tid]; DGWs[tid] = dgw[tid]; }

  // sc tile: [ea(4) | xs_src(6) | xs_dst(6) | 0-pad(16)]
  {
    int e = e0 + (tid >> 1);
    int part = tid & 1;
    half_t* row = &SC[(tid >> 1) * GT_LDK];
    if (part == 0) {
      float t[16];
      if (e < E) {
        int s = src[e], d = dst[e];
        float4 e4 = *(const float4*)(ea + (size_t)e * 4);
        const float* ps = xs_s + (size_t)s * 6;
        const float* pd = xs_d + (size_t)d * 6;
        t[0] = e4.x; t[1] = e4.y; t[2] = e4.z; t[3] = e4.w;
#pragma unroll
        for (int k = 0; k < 6; ++k) t[4 + k] = ps[k];
#pragma unroll
        for (int k = 0; k < 6; ++k) t[10 + k] = pd[k];
      } else {
#pragma unroll
        for (int k = 0; k < 16; ++k) t[k] = 0.f;
      }
      v8h h0, h1;
#pragma unroll
      for (int i = 0; i < 8; ++i) { h0[i] = (half_t)t[i]; h1[i] = (half_t)t[8 + i]; }
      *(v8h*)(row)     = h0;
      *(v8h*)(row + 8) = h1;
    } else {
      v8h z;
#pragma unroll
      for (int i = 0; i < 8; ++i) z[i] = (half_t)0.f;
      *(v8h*)(row + 16) = z;
      *(v8h*)(row + 24) = z;
    }
  }
  __syncthreads();

  {
    v16h a = load_a_frag(SC, wave * 16, GT_LDK, lane);
    float pb[8];
#pragma unroll
    for (int j = 0; j < 8; ++j) pb[j] = 0.f;
#pragma unroll
    for (int nf = 0; nf < 8; ++nf) {
      v16h b = load_b_frag(WS, nf * 16, GT_LDK, lane);
      v8f acc = {};
      acc = wmma32(a, b, acc);
      int n = nf * 16 + (lane & 15);
      float bb = B1s[n], cwn = CWs[n];
#pragma unroll
      for (int j = 0; j < 8; ++j) {
        float u = fmaxf(acc[j] + bb, 0.f);     // relu(fc1)
        pb[j] = fmaf(u, cwn, pb[j]);           // · (bw·W2)
      }
    }
#pragma unroll
    for (int j = 0; j < 8; ++j) {
      float s = pb[j];
      s += __shfl_xor(s, 8, 16);
      s += __shfl_xor(s, 4, 16);
      s += __shfl_xor(s, 2, 16);
      s += __shfl_xor(s, 1, 16);
      pb[j] = s;
    }
    if ((lane & 15) == 0) {
      int rbase = wave * 16 + (lane >> 4) * 8;
#pragma unroll
      for (int j = 0; j < 8; ++j) BE[rbase + j] = pb[j];
    }
  }
  __syncthreads();

  if (tid < 128) {
    int e = e0 + tid;
    if (e < E) {
      float be = BE[tid] + cw[128];
      be = (be > 20.f) ? be : log1pf(__expf(be));           // softplus
      int s = src[e], d = dst[e];
      const half_t* gs = gsrc + (size_t)s * 128;
      const half_t* gd = gdst + (size_t)d * 128;
      float acc = 0.f;
#pragma unroll 4
      for (int k8 = 0; k8 < 16; ++k8) {
        v8h a8 = *(const v8h*)(gs + k8 * 8);
        v8h b8 = *(const v8h*)(gd + k8 * 8);
#pragma unroll
        for (int i = 0; i < 8; ++i) {
          float g = (float)a8[i] + (float)b8[i];
          acc = fmaf(fmaxf(g, 0.f), DGWs[k8 * 8 + i], acc);
        }
      }
      float ge = 1.f / (1.f + __expf(-(acc + dgb[0])));
      COEF[tid] = be * ge;
    }
  }
  __syncthreads();

  {
    int e = e0 + (tid >> 1);
    if (e < E) {
      int part = tid & 1;
      float c = COEF[tid >> 1];
      int s = src[e], d = dst[e];
      const float* vp = v + (size_t)s * 64 + part * 32;
      float* mp = msg + (size_t)d * 64 + part * 32;
#pragma unroll 8
      for (int i = 0; i < 32; ++i) atomicAdd(&mp[i], c * vp[i]);
    }
  }
}

// ---------------------------------------------------------------------------
// GAT kernels
// ---------------------------------------------------------------------------
__device__ inline void atomicMaxFloat(float* addr, float val) {
  if (val >= 0.f) atomicMax((int*)addr, __float_as_int(val));
  else            atomicMin((unsigned int*)addr, (unsigned int)__float_as_int(val));
}

__global__ __launch_bounds__(256)
void gat_pass1_kernel(const int* __restrict__ src, const int* __restrict__ dst, int E,
                      const float* __restrict__ xl, const float* __restrict__ xr,
                      const float* __restrict__ att,
                      float* __restrict__ e_out, float* __restrict__ mmax)
{
  __shared__ float Att[64];
  if (threadIdx.x < 64) Att[threadIdx.x] = att[threadIdx.x];
  __syncthreads();
  int e = blockIdx.x * 256 + threadIdx.x;
  if (e >= E) return;
  int s = src[e], d = dst[e];
  const float* pl = xl + (size_t)s * 64;
  const float* pr = xr + (size_t)d * 64;
  float4 eh;
  float* ehp = (float*)&eh;
#pragma unroll
  for (int h = 0; h < 4; ++h) {
    float acc = 0.f;
#pragma unroll
    for (int c = 0; c < 16; ++c) {
      float x = pl[h * 16 + c] + pr[h * 16 + c];
      x = (x > 0.f) ? x : 0.2f * x;                         // leaky_relu 0.2
      acc = fmaf(x, Att[h * 16 + c], acc);
    }
    ehp[h] = acc;
    atomicMaxFloat(&mmax[(size_t)d * 4 + h], acc);
  }
  *(float4*)(e_out + (size_t)e * 4) = eh;
}

__global__ __launch_bounds__(256)
void gat_pass2_kernel(const int* __restrict__ src, const int* __restrict__ dst, int E,
                      const float* __restrict__ xl, const float* __restrict__ e_in,
                      const float* __restrict__ mmax,
                      float* __restrict__ den, float* __restrict__ num)
{
  int e = blockIdx.x * 256 + threadIdx.x;
  if (e >= E) return;
  int s = src[e], d = dst[e];
  const float* pl = xl + (size_t)s * 64;
  float* pn = num + (size_t)d * 64;
  float4 eh = *(const float4*)(e_in + (size_t)e * 4);
  float4 mh = *(const float4*)(mmax + (size_t)d * 4);
  const float* ehp = (const float*)&eh;
  const float* mhp = (const float*)&mh;
#pragma unroll
  for (int h = 0; h < 4; ++h) {
    float ex = __expf(ehp[h] - mhp[h]);
    atomicAdd(&den[(size_t)d * 4 + h], ex);
#pragma unroll
    for (int c = 0; c < 16; ++c)
      atomicAdd(&pn[h * 16 + c], ex * pl[h * 16 + c]);
  }
}

__global__ __launch_bounds__(256)
void gat_finish_kernel(int N, const float* __restrict__ num, const float* __restrict__ den,
                       const float* __restrict__ res, float* __restrict__ pre)
{
  int i = blockIdx.x * 256 + threadIdx.x;
  if (i >= N * 64) return;
  int n = i >> 6, h = (i & 63) >> 4;
  pre[i] = num[i] / (den[(size_t)n * 4 + h] + 1e-16f) + res[i];
}

// ---------------------------------------------------------------------------
// elementwise kernels
// ---------------------------------------------------------------------------
__global__ void fill_kernel(float* __restrict__ p, float v, size_t n) {
  size_t i = (size_t)blockIdx.x * 256 + threadIdx.x;
  if (i < n) p[i] = v;
}

__global__ __launch_bounds__(256)
void ln_kernel(const float* __restrict__ in, int istride, int ioff,
               float* __restrict__ out, int ostride, int ooff,
               const float* __restrict__ g, const float* __restrict__ b,
               int D, int rows)
{
  int gt = blockIdx.x * 256 + threadIdx.x;
  int row = gt >> 5, lane = gt & 31;
  if (row >= rows) return;
  const float* x = in + (size_t)row * istride + ioff;
  float v0 = (lane      < D) ? x[lane]      : 0.f;
  float v1 = (lane + 32 < D) ? x[lane + 32] : 0.f;
  float v2 = (lane + 64 < D) ? x[lane + 64] : 0.f;
  float s = v0 + v1 + v2;
#pragma unroll
  for (int o = 16; o >= 1; o >>= 1) s += __shfl_xor(s, o, 32);
  float mean = s / (float)D;
  float d0 = v0 - mean, d1 = v1 - mean, d2 = v2 - mean;
  float q = 0.f;
  if (lane      < D) q += d0 * d0;
  if (lane + 32 < D) q += d1 * d1;
  if (lane + 64 < D) q += d2 * d2;
#pragma unroll
  for (int o = 16; o >= 1; o >>= 1) q += __shfl_xor(q, o, 32);
  float inv = rsqrtf(q / (float)D + 1e-5f);
  float* y = out + (size_t)row * ostride + ooff;
  if (lane      < D) y[lane]      = d0 * inv * g[lane]      + b[lane];
  if (lane + 32 < D) y[lane + 32] = d1 * inv * g[lane + 32] + b[lane + 32];
  if (lane + 64 < D) y[lane + 64] = d2 * inv * g[lane + 64] + b[lane + 64];
}

__global__ void gelu_res_kernel(const float* __restrict__ h, const float* __restrict__ p,
                                float* __restrict__ out, size_t n)
{
  size_t i = (size_t)blockIdx.x * 256 + threadIdx.x;
  if (i >= n) return;
  float x = p[i];
  out[i] = h[i] + 0.5f * x * (1.f + erff(x * 0.70710678118654752f));
}

__global__ void gru_kernel(int N, const float* __restrict__ gi, const float* __restrict__ gh,
                           const float* __restrict__ h, float* __restrict__ out)
{
  int i = blockIdx.x * 256 + threadIdx.x;
  if (i >= N * 96) return;
  int n = i / 96, j = i - n * 96;
  const float* gin = gi + (size_t)n * 288;
  const float* ghn = gh + (size_t)n * 288;
  float r  = 1.f / (1.f + __expf(-(gin[j]       + ghn[j])));
  float z  = 1.f / (1.f + __expf(-(gin[96 + j]  + ghn[96 + j])));
  float ng = tanhf(gin[192 + j] + r * ghn[192 + j]);
  out[i] = (1.f - z) * ng + z * h[i];
}

// ---------------------------------------------------------------------------
// host side: param pytree walking (JAX tree_flatten order) + orchestration
// ---------------------------------------------------------------------------
struct Lin  { const float* b; const float* w; };
struct Mlp  { Lin fc1, fc2; };
struct LNp  { const float* b; const float* g; };
struct SDp  { Lin bw; Mlp dg; Mlp pl; Mlp se; };
struct GATp { const float* att; const float* bias; Mlp ffn; Lin lin_l; Lin lin_r; const float* res_w; };
struct InterP { LNp h_ln, msg_ln; Lin proj; };
struct NodeP  { LNp dyn_ln; Lin dyn_proj; const float* b_hh; const float* b_ih;
                const float* w_hh; const float* w_ih; LNp h_ln, msg_ln; };
struct RoundP { SDp oo; GATp ot; GATp to; SDp tt; };

struct PReader {
  void* const* din; int idx; const float* flat; size_t off; bool leafmode;
  const float* next(size_t n) {
    if (leafmode) return (const float*)din[idx++];
    const float* p = flat + off; off += n; return p;
  }
};

static inline int aligned16(const void* p) { return ((uintptr_t)p & 15) == 0 ? 1 : 0; }

extern "C" void kernel_launch(void* const* d_in, const int* in_sizes, int n_in,
                              void* d_out, int out_size, void* d_ws, size_t ws_size,
                              hipStream_t stream)
{
  (void)in_sizes; (void)out_size; (void)ws_size;
  const int N1 = N1_C, N2 = N2_C;
  const float* h1_in  = (const float*)d_in[0];
  const float* h2_in  = (const float*)d_in[1];
  const float* xd1    = (const float*)d_in[2];
  const float* xd2    = (const float*)d_in[3];
  const float* xs1    = (const float*)d_in[4];
  const float* xs2    = (const float*)d_in[5];
  const float* ea_oo  = (const float*)d_in[6];
  const float* ea_tt  = (const float*)d_in[7];
  const int* ei_oo    = (const int*)d_in[8];
  const int* ei_tt    = (const int*)d_in[9];
  const int* ei_ot    = (const int*)d_in[10];
  const int* ei_to    = (const int*)d_in[11];

  PReader pr{};
  if (n_in > 13) { pr.leafmode = true;  pr.din = d_in; pr.idx = 12; }
  else           { pr.leafmode = false; pr.flat = (const float*)d_in[12]; pr.off = 0; }

  auto getLin = [&](int od, int id) { Lin l; l.b = pr.next(od); l.w = pr.next((size_t)od * id); return l; };
  auto getMlp = [&](int id, int hid, int od) { Mlp m; m.fc1 = getLin(hid, id); m.fc2 = getLin(od, hid); return m; };
  auto getLN  = [&](int d) { LNp l; l.b = pr.next(d); l.g = pr.next(d); return l; };
  auto getSD  = [&]() { SDp s; s.bw = getLin(1, 128); s.dg = getMlp(192, 128, 1);
                        s.pl = getMlp(96, 128, 64); s.se = getMlp(16, 128, 128); return s; };
  auto getGAT = [&]() { GATp g; g.att = pr.next(64); g.bias = pr.next(64);
                        g.ffn = getMlp(64, 128, 64); g.lin_l = getLin(64, 96);
                        g.lin_r = getLin(64, 96); g.res_w = pr.next(64 * 96); return g; };

  // params flatten order: "inter" -> "node" -> "rounds" (keys sorted per level)
  InterP inter1[2];
  for (int t = 0; t < 2; ++t) {
    inter1[t].h_ln = getLN(96); inter1[t].msg_ln = getLN(64); inter1[t].proj = getLin(96, 64);
  }
  NodeP node[2];
  for (int t = 0; t < 2; ++t) {
    node[t].dyn_ln = getLN(64); node[t].dyn_proj = getLin(64, 8);
    node[t].b_hh = pr.next(288); node[t].b_ih = pr.next(288);
    node[t].w_hh = pr.next((size_t)288 * 96); node[t].w_ih = pr.next((size_t)288 * 128);
    node[t].h_ln = getLN(96); node[t].msg_ln = getLN(64);
  }
  RoundP rounds[2];
  for (int r = 0; r < 2; ++r) {
    rounds[r].oo = getSD(); rounds[r].ot = getGAT(); rounds[r].to = getGAT(); rounds[r].tt = getSD();
  }

  // workspace carve
  char* wp = (char*)d_ws; size_t off = 0;
  auto carve = [&](size_t bytes) { void* p = wp + off; off = (off + bytes + 255) & ~(size_t)255; return p; };
  float* hcur1 = (float*)carve((size_t)N1 * 96 * 4);
  float* hcur2 = (float*)carve((size_t)N2 * 96 * 4);
  float* msg1  = (float*)carve((size_t)N1 * 64 * 4);
  float* msg2  = (float*)carve((size_t)N2 * 64 * 4);
  const size_t BIG = (size_t)N2 * 288 * 4;
  float* bufA  = (float*)carve(BIG);
  float* bufB  = (float*)carve(BIG);
  float* bufC  = (float*)carve(BIG);
  float* bufD  = (float*)carve((size_t)N2 * 64 * 4);
  float* cwbuf = (float*)carve(1024);

  auto fill = [&](float* p, float v, size_t n) {
    fill_kernel<<<(unsigned)((n + 255) / 256), 256, 0, stream>>>(p, v, n);
  };
  auto gemm = [&](const float* A, int M, int K, int lda,
                  const float* W, int ldw, int wcol0, const float* bias,
                  float* C, half_t* Ch, int ldc, int ccol0, int Nout, int act, int accum) {
    dim3 grid((M + 63) / 64, (Nout + 63) / 64);
    gemm_kernel<<<grid, 256, 0, stream>>>(A, M, K, lda, W, ldw, wcol0, bias, C, Ch,
                                          ldc, ccol0, Nout, act, accum, aligned16(W));
  };
  auto lnrun = [&](const float* in, int is, int io, float* o, int os, int oo2,
                   const LNp& l, int D, int rows) {
    ln_kernel<<<(rows + 7) / 8, 256, 0, stream>>>(in, is, io, o, os, oo2, l.g, l.b, D, rows);
  };

  auto sd_pass = [&](const SDp& p, const float* h, const float* xs, int N,
                     const float* ea, const int* ei, int E, float* msg) {
    sd_prep_kernel<<<1, 128, 0, stream>>>(p.bw.w, p.bw.b, p.se.fc2.w, p.se.fc2.b, cwbuf);
    // dynamic-gate node GEMMs (f16 outputs; fc1 bias folded into src half)
    gemm(h, N, 96, 96, p.dg.fc1.w, 192, 0,  p.dg.fc1.b, nullptr, (half_t*)bufB, 128, 0, 128, 0, 0);
    gemm(h, N, 96, 96, p.dg.fc1.w, 192, 96, nullptr,    nullptr, (half_t*)bufC, 128, 0, 128, 0, 0);
    // payload MLP on nodes
    gemm(h, N, 96, 96, p.pl.fc1.w, 96, 0, p.pl.fc1.b, bufA, nullptr, 128, 0, 128, 1, 0);
    gemm(bufA, N, 128, 128, p.pl.fc2.w, 128, 0, p.pl.fc2.b, bufD, nullptr, 64, 0, 64, 0, 0);
    sd_edge_kernel<<<(E + 127) / 128, 256, 0, stream>>>(ei, ei + E, E, ea, xs, xs,
        p.se.fc1.w, p.se.fc1.b, cwbuf, (const half_t*)bufB, (const half_t*)bufC,
        p.dg.fc2.w, p.dg.fc2.b, bufD, msg, aligned16(p.se.fc1.w));
  };

  auto gat_pass = [&](const GATp& p, const float* hs, int Ns, const float* hd, int Nd,
                      const int* ei, int E, float* msg) {
    float* xl  = bufA;
    float* xr  = bufD;
    float* ews = bufB;
    float* m   = bufC;
    float* den = bufC + (size_t)N2 * 4;
    float* num = bufC + (size_t)N2 * 8;
    float* res = bufB + (size_t)E_TT_C * 4;
    gemm(hs, Ns, 96, 96, p.lin_l.w, 96, 0, p.lin_l.b, xl, nullptr, 64, 0, 64, 0, 0);
    gemm(hd, Nd, 96, 96, p.lin_r.w, 96, 0, p.lin_r.b, xr, nullptr, 64, 0, 64, 0, 0);
    fill(m, -3.0e38f, (size_t)Nd * 4);
    fill(den, 0.f, (size_t)Nd * 4);
    fill(num, 0.f, (size_t)Nd * 64);
    int eb = (E + 255) / 256;
    gat_pass1_kernel<<<eb, 256, 0, stream>>>(ei, ei + E, E, xl, xr, p.att, ews, m);
    gat_pass2_kernel<<<eb, 256, 0, stream>>>(ei, ei + E, E, xl, ews, m, den, num);
    gemm(hd, Nd, 96, 96, p.res_w, 96, 0, p.bias, res, nullptr, 64, 0, 64, 0, 0);
    gat_finish_kernel<<<(Nd * 64 + 255) / 256, 256, 0, stream>>>(Nd, num, den, res, xr);
    gemm(xr, Nd, 64, 64, p.ffn.fc1.w, 64, 0, p.ffn.fc1.b, xl, nullptr, 128, 0, 128, 1, 0);
    gemm(xl, Nd, 128, 128, p.ffn.fc2.w, 128, 0, p.ffn.fc2.b, msg, nullptr, 64, 0, 64, 0, 1);
  };

  const float* h1 = h1_in;
  const float* h2 = h2_in;
  for (int r = 0; r < 2; ++r) {
    fill(msg1, 0.f, (size_t)N1 * 64);
    fill(msg2, 0.f, (size_t)N2 * 64);
    sd_pass(rounds[r].oo, h1, xs1, N1, ea_oo, ei_oo, E_OO_C, msg1);
    gat_pass(rounds[r].to, h2, N2, h1, N1, ei_to, E_TO_C, msg1);
    sd_pass(rounds[r].tt, h2, xs2, N2, ea_tt, ei_tt, E_TT_C, msg2);
    gat_pass(rounds[r].ot, h1, N1, h2, N2, ei_ot, E_OT_C, msg2);
    if (r == 0) {
      for (int t = 0; t < 2; ++t) {
        float* msg = (t == 0) ? msg1 : msg2;
        const float* h = (t == 0) ? h1 : h2;
        float* hcur = (t == 0) ? hcur1 : hcur2;
        int N = (t == 0) ? N1 : N2;
        lnrun(msg, 64, 0, bufB, 64, 0, inter1[t].msg_ln, 64, N);
        gemm(bufB, N, 64, 64, inter1[t].proj.w, 64, 0, inter1[t].proj.b, bufA, nullptr, 96, 0, 96, 0, 0);
        gelu_res_kernel<<<(unsigned)(((size_t)N * 96 + 255) / 256), 256, 0, stream>>>(h, bufA, bufC, (size_t)N * 96);
        lnrun(bufC, 96, 0, hcur, 96, 0, inter1[t].h_ln, 96, N);
      }
      h1 = hcur1; h2 = hcur2;
    }
  }

  // final GRU update per node type
  for (int t = 0; t < 2; ++t) {
    float* msg = (t == 0) ? msg1 : msg2;
    const float* h = (t == 0) ? h1 : h2;
    const float* xd = (t == 0) ? xd1 : xd2;
    int N = (t == 0) ? N1 : N2;
    size_t rowoff = (t == 0) ? 0 : (size_t)N1;
    // xcat = [ LN(dyn) , LN(msg) ]  in bufB with row stride 128
    gemm(xd, N, 8, 8, node[t].dyn_proj.w, 8, 0, node[t].dyn_proj.b, bufB, nullptr, 128, 0, 64, 0, 0);
    lnrun(bufB, 128, 0, bufB, 128, 0, node[t].dyn_ln, 64, N);
    lnrun(msg, 64, 0, bufB, 128, 64, node[t].msg_ln, 64, N);
    gemm(bufB, N, 128, 128, node[t].w_ih, 128, 0, node[t].b_ih, bufA, nullptr, 288, 0, 288, 0, 0);
    gemm(h, N, 96, 96, node[t].w_hh, 96, 0, node[t].b_hh, bufC, nullptr, 288, 0, 288, 0, 0);
    gru_kernel<<<(unsigned)(((size_t)N * 96 + 255) / 256), 256, 0, stream>>>(N, bufA, bufC, h, bufB);
    lnrun(bufB, 96, 0, (float*)d_out + rowoff * 96, 96, 0, node[t].h_ln, 96, N);
  }
}
